// CustomMultiHeadAttentionGPT_79723182948668
// MI455X (gfx1250) — compile-verified, matrix operand order checked by reading, not run on hardware
//
#include <hip/hip_runtime.h>
#include <stdint.h>

#define B_    4
#define T_    1024
#define D_    1024
#define H_    16
#define L_    6
#define V_    50257
#define VPAD_ 50258
#define NM_   2
#define DH_   64
#define DFF_  4096
#define BT_   (B_*T_)

#define USE_ASYNC 1
#define USE_TDM   1

typedef __attribute__((ext_vector_type(16))) __bf16 v16bf;
typedef __attribute__((ext_vector_type(8)))  float  v8f;
typedef __attribute__((ext_vector_type(4)))  uint32_t v4u;
typedef __attribute__((ext_vector_type(8)))  int32_t  v8i;
typedef __attribute__((ext_vector_type(4)))  int32_t  v4i;

__device__ __forceinline__ uint16_t f2bf(float f) {
  uint32_t u = __float_as_uint(f);
  return (uint16_t)((u + 0x7FFFu + ((u >> 16) & 1u)) >> 16);
}

union FragU { v16bf v; uint32_t u[8]; };

// LDS byte offset of a __shared__ object: low 32 bits of the flat shared
// aperture address are the LDS offset (ISA 10.2 per-aperture mapping).
__device__ __forceinline__ uint32_t lds_off(const void* p) {
  return (uint32_t)(uintptr_t)p;
}

#if USE_ASYNC
// GLOBAL_LOAD_ASYNC_TO_LDS_B128: per-lane 16B global -> LDS, ASYNCcnt-tracked.
__device__ __forceinline__ void async_ld_b128(uint32_t lds, const void* gptr) {
  asm volatile("global_load_async_to_lds_b128 %0, %1, off"
               :: "v"(lds), "v"((uint64_t)(uintptr_t)gptr) : "memory");
}
__device__ __forceinline__ void wait_async0() {
  asm volatile("s_wait_asynccnt 0" ::: "memory");
}
#endif

#if USE_TDM
// Tensor Data Mover 2D tile load: global (strided rows) -> LDS (contiguous).
// D# per ISA 8.3/8.4: count=1, type=2 (image), data_size=2B, iterate/pad/gather
// off, huge tensor dims (tiles are always fully in-bounds -> no OOB zeroing).
// This toolchain (clang-23 / therock headers) uses the 6-arg builtin form:
// (uint32x4 g0, int32x8 g1, int32x4 g2, int32x4 g3, int32x8, i32 cpol).
__device__ __forceinline__ void tdm_load_2d(uint32_t lds, const void* gptr,
                                            uint32_t tile0, uint32_t tile1,
                                            uint64_t stride0 /*elements*/) {
  uint64_t ga = (uint64_t)(uintptr_t)gptr;
  const uint32_t dim0 = 0x7FFFFFFFu, dim1 = 0x7FFFFFFFu;
  v4u g0;
  g0[0] = 1u;                                             // count=1, user mode
  g0[1] = lds;                                            // lds_addr (bytes)
  g0[2] = (uint32_t)ga;                                   // global_addr[31:0]
  g0[3] = (uint32_t)((ga >> 32) & 0x1FFFFFFu) | (2u << 30);  // addr[56:32], type=2
  v8i g1;
  g1[0] = (int32_t)(1u << 16);                            // data_size = 2 bytes
  g1[1] = (int32_t)((dim0 & 0xFFFFu) << 16);              // tensor_dim0[15:0]
  g1[2] = (int32_t)(((dim0 >> 16) & 0xFFFFu) | ((dim1 & 0xFFFFu) << 16));
  g1[3] = (int32_t)(((dim1 >> 16) & 0xFFFFu) | (tile0 << 16));   // tile_dim0
  g1[4] = (int32_t)(tile1 & 0xFFFFu);                     // tile_dim1; tile_dim2=0
  g1[5] = (int32_t)(uint32_t)(stride0 & 0xFFFFFFFFu);     // dim0_stride[31:0]
  g1[6] = (int32_t)(uint32_t)((stride0 >> 32) & 0xFFFFu); // dim0_stride[47:32]
  g1[7] = 0;
  v4i gz4 = {0, 0, 0, 0};
  v8i gz8 = {0, 0, 0, 0, 0, 0, 0, 0};
  __builtin_amdgcn_tensor_load_to_lds(g0, g1, gz4, gz4, gz8, 0);
}
#endif

__device__ __forceinline__ void wait_stage() {
#if USE_TDM
  __builtin_amdgcn_s_wait_tensorcnt(0);
#endif
#if USE_ASYNC
  wait_async0();
#endif
}

// ---------------------------------------------------------------------------
// Tiled bf16 WMMA GEMM. C[M,N] = A[M,K] * B + bias, optional GELU, +resid.
// BTRANS=false: B is [K,N] row-major.  BTRANS=true: B is [N,K] row-major (A*B^T).
// Block = 256 threads = 8 waves; tile 256x128, K-step 32, double-buffered LDS.
// Wave w: (wm = w>>1) x (wn = w&1) -> 64x64 region = 4x4 WMMA 16x16 tiles.
// A and NT-B tiles move via TDM (wave 0, TENSORcnt); NN-B transposes through
// registers into LDS. B always lands as Bt[n][k] so fragment loads are
// uniform aligned ds_load_b128 pairs.
// M must be a multiple of 256, K a multiple of 32; N is bounds-checked.
// ---------------------------------------------------------------------------
template<bool BTRANS>
__global__ __launch_bounds__(256)
void gemm_bf16_k(const uint16_t* __restrict__ A, const uint16_t* __restrict__ Bm,
                 const float* __restrict__ bias, const float* __restrict__ resid,
                 float* __restrict__ Cf, uint16_t* __restrict__ Cb,
                 int M, int N, int K, int lda, int ldb, int ldc,
                 long long sAb, long long sAh, long long sBb, long long sBh,
                 long long sCb, long long sCh, int Hdim, int doGelu)
{
  __shared__ __align__(16) uint16_t As[2][256*32];
  __shared__ __align__(16) uint16_t Bs[2][128*32];   // Bt[n][k] layout

  int z  = blockIdx.z;
  int bi = z / Hdim, hi = z % Hdim;
  A  += (size_t)bi*sAb + (size_t)hi*sAh;
  Bm += (size_t)bi*sBb + (size_t)hi*sBh;
  size_t cOff = (size_t)bi*sCb + (size_t)hi*sCh;
  if (Cf)    Cf    += cOff;
  if (Cb)    Cb    += cOff;
  if (resid) resid += cOff;

  int t    = threadIdx.x;
  int lane = t & 31, w = t >> 5;
  int wm   = w >> 1, wn = w & 1;
  int lr   = lane & 15, kh = lane >> 4;
  int mBase = blockIdx.y * 256;
  int nBase = blockIdx.x * 128;

  v8f zero = {};
  v8f acc[4][4];
#pragma unroll
  for (int i = 0; i < 4; i++)
#pragma unroll
    for (int j = 0; j < 4; j++) acc[i][j] = zero;

  // K offsets per fragment dword (ISA 7.12.2 16-bit A/B layout):
  // lanes 0-15 hold K={0..7,16..23}, lanes 16-31 hold K={8..15,24..31}.
  int kk[8];
#pragma unroll
  for (int g = 0; g < 8; g++) kk[g] = 2*g + (g >= 4 ? 8 : 0) + 8*kh;

  auto stageA = [&](int buf, int k0) {
#if USE_TDM
    if (w == 0)
      tdm_load_2d(lds_off(&As[buf][0]), A + (size_t)mBase * lda + k0,
                  32u, 256u, (uint64_t)lda);
#else
    int r = t;  // 256 rows, one per thread
    const uint16_t* ga = A + (size_t)(mBase + r) * lda + k0;
    uint32_t la = lds_off(&As[buf][r*32]);
#if USE_ASYNC
    async_ld_b128(la,      ga);
    async_ld_b128(la + 16, ga + 8);
    async_ld_b128(la + 32, ga + 16);
    async_ld_b128(la + 48, ga + 24);
#else
    const uint32_t* s = (const uint32_t*)ga;
    uint32_t* d = (uint32_t*)&As[buf][r*32];
#pragma unroll
    for (int i = 0; i < 16; i++) d[i] = s[i];
#endif
#endif
  };

  auto stageB = [&](int buf, int k0) {
    if (BTRANS) {
      if (nBase + 128 <= N) {
#if USE_TDM
        if (w == 0)
          tdm_load_2d(lds_off(&Bs[buf][0]), Bm + (size_t)nBase * ldb + k0,
                      32u, 128u, (uint64_t)ldb);
#else
        int r = t >> 1, c0 = (t & 1) * 16;
        const uint16_t* gb = Bm + (size_t)(nBase + r) * ldb + k0 + c0;
        uint32_t lb = lds_off(&Bs[buf][r*32 + c0]);
#if USE_ASYNC
        async_ld_b128(lb,      gb);
        async_ld_b128(lb + 16, gb + 8);
#else
        const uint32_t* s = (const uint32_t*)gb;
        uint32_t* d = (uint32_t*)&Bs[buf][r*32 + c0];
#pragma unroll
        for (int i = 0; i < 8; i++) d[i] = s[i];
#endif
#endif
      } else {
        int r = t >> 1, c0 = (t & 1) * 16;
        int gn = nBase + r;
        uint32_t* d = (uint32_t*)&Bs[buf][r*32 + c0];
        if (gn < N) {
          const uint32_t* s = (const uint32_t*)(Bm + (size_t)gn * ldb + k0 + c0);
#pragma unroll
          for (int i = 0; i < 8; i++) d[i] = s[i];
        } else {
#pragma unroll
          for (int i = 0; i < 8; i++) d[i] = 0u;
        }
      }
    } else {
      // B row-major [K,N]: transpose while staging.
      // Thread: k-pair kp = t&15 (k = 2kp, 2kp+1), cols n0 = (t>>4)*8.
      int kp = t & 15, n0 = (t >> 4) * 8;
      int k  = 2 * kp;
      if (((ldb & 1) == 0) && (nBase + 128 <= N)) {
        const uint32_t* r0 = (const uint32_t*)(Bm + (size_t)(k0 + k    ) * ldb + nBase + n0);
        const uint32_t* r1 = (const uint32_t*)(Bm + (size_t)(k0 + k + 1) * ldb + nBase + n0);
        uint32_t a0[4], a1[4];
#pragma unroll
        for (int i = 0; i < 4; i++) { a0[i] = r0[i]; a1[i] = r1[i]; }
#pragma unroll
        for (int i = 0; i < 4; i++) {
          uint32_t o0 = (a0[i] & 0xFFFFu) | (a1[i] << 16);            // cols n0+2i
          uint32_t o1 = (a0[i] >> 16)     | (a1[i] & 0xFFFF0000u);    // cols n0+2i+1
          *(uint32_t*)(&Bs[buf][(n0 + 2*i    ) * 32 + k]) = o0;
          *(uint32_t*)(&Bs[buf][(n0 + 2*i + 1) * 32 + k]) = o1;
        }
        if (k0 + 32 < K)
          __builtin_prefetch(Bm + (size_t)(k0 + 32 + k) * ldb + nBase + n0, 0, 3);
      } else {
#pragma unroll
        for (int j = 0; j < 8; j++) {
          int gn = nBase + n0 + j;
          uint32_t lo = 0, hi = 0;
          if (gn < N) {
            lo = Bm[(size_t)(k0 + k    ) * ldb + gn];
            hi = Bm[(size_t)(k0 + k + 1) * ldb + gn];
          }
          *(uint32_t*)(&Bs[buf][(n0 + j) * 32 + k]) = lo | (hi << 16);
        }
      }
    }
  };

  // ---- double-buffered main loop ----
  int nk = K / 32;
  stageA(0, 0);
  stageB(0, 0);
  wait_stage();
  __syncthreads();

  for (int ki = 0; ki < nk; ki++) {
    int cur = ki & 1;
    if (ki + 1 < nk) {            // issue next tile while computing this one
      stageA(cur ^ 1, (ki + 1) * 32);
      stageB(cur ^ 1, (ki + 1) * 32);
    }

    FragU a[4], b[4];
#pragma unroll
    for (int ms = 0; ms < 4; ms++) {
      int row = wm*64 + ms*16 + lr;
#pragma unroll
      for (int g = 0; g < 8; g++)
        a[ms].u[g] = *(const uint32_t*)(&As[cur][row*32 + kk[g]]);
    }
#pragma unroll
    for (int ns = 0; ns < 4; ns++) {
      int col = wn*64 + ns*16 + lr;
#pragma unroll
      for (int g = 0; g < 8; g++)
        b[ns].u[g] = *(const uint32_t*)(&Bs[cur][col*32 + kk[g]]);
    }
#pragma unroll
    for (int ms = 0; ms < 4; ms++)
#pragma unroll
      for (int ns = 0; ns < 4; ns++)
        acc[ms][ns] = __builtin_amdgcn_wmma_f32_16x16x32_bf16(
            false, a[ms].v, false, b[ns].v, (short)0, acc[ms][ns], false, false);

    wait_stage();                 // next tile landed (tensorcnt + asynccnt)
    __syncthreads();              // all waves done reading `cur`
  }

  // ---- epilogue: bias / gelu / residual / f32 and bf16 stores ----
#pragma unroll
  for (int ms = 0; ms < 4; ms++) {
#pragma unroll
    for (int ns = 0; ns < 4; ns++) {
      int col = nBase + wn*64 + ns*16 + lr;
      if (col >= N) continue;
      float bv = bias ? bias[col] : 0.0f;
#pragma unroll
      for (int i = 0; i < 8; i++) {
        int row = mBase + wm*64 + ms*16 + kh*8 + i;
        float v = acc[ms][ns][i] + bv;
        if (doGelu) v = 0.5f * v * (1.0f + erff(v * 0.70710678118654752f));
        if (resid)  v += resid[(size_t)row * ldc + col];
        if (Cf) Cf[(size_t)row * ldc + col] = v;
        if (Cb) Cb[(size_t)row * ldc + col] = f2bf(v);
      }
    }
  }
}

// ---------------------------------------------------------------------------
// LayerNorm: one block per row of D=1024, output bf16.
// ---------------------------------------------------------------------------
__global__ __launch_bounds__(256)
void layernorm_k(const float* __restrict__ x, const float* __restrict__ w,
                 const float* __restrict__ b, uint16_t* __restrict__ out)
{
  __shared__ float s1[256], s2[256];
  int r = blockIdx.x;
  const float* xr = x + (size_t)r * D_;
  int t = threadIdx.x;
  float vals[4];
  float lsum = 0.f, lsq = 0.f;
#pragma unroll
  for (int i = 0; i < 4; i++) {
    float v = xr[t + i*256];
    vals[i] = v; lsum += v; lsq += v*v;
  }
  s1[t] = lsum; s2[t] = lsq; __syncthreads();
  for (int off = 128; off > 0; off >>= 1) {
    if (t < off) { s1[t] += s1[t+off]; s2[t] += s2[t+off]; }
    __syncthreads();
  }
  float mu  = s1[0] * (1.0f / D_);
  float var = s2[0] * (1.0f / D_) - mu*mu;
  float rs  = rsqrtf(var + 1e-5f);
#pragma unroll
  for (int i = 0; i < 4; i++) {
    int c = t + i*256;
    out[(size_t)r * D_ + c] = f2bf((vals[i] - mu) * rs * w[c] + b[c]);
  }
}

// ---------------------------------------------------------------------------
// Masked softmax over T=1024: p = softmax(scores*scale*mask), bf16 out.
// ---------------------------------------------------------------------------
__global__ __launch_bounds__(256)
void softmax_mask_k(const float* __restrict__ scores, const float* __restrict__ mask,
                    uint16_t* __restrict__ probs, float scale)
{
  __shared__ float sh[256];
  size_t r = blockIdx.x;
  int q = (int)(r % T_);
  const float* sr = scores + r * (size_t)T_;
  const float* mr = mask   + (size_t)q * T_;
  int t = threadIdx.x;
  float v[4];
  float lmax = -1e30f;
#pragma unroll
  for (int i = 0; i < 4; i++) {
    int c = t + i*256;
    float x = sr[c] * scale * mr[c];
    v[i] = x; lmax = fmaxf(lmax, x);
  }
  sh[t] = lmax; __syncthreads();
  for (int off = 128; off > 0; off >>= 1) {
    if (t < off) sh[t] = fmaxf(sh[t], sh[t+off]);
    __syncthreads();
  }
  float m = sh[0]; __syncthreads();
  float lsum = 0.f;
#pragma unroll
  for (int i = 0; i < 4; i++) { v[i] = expf(v[i] - m); lsum += v[i]; }
  sh[t] = lsum; __syncthreads();
  for (int off = 128; off > 0; off >>= 1) {
    if (t < off) sh[t] += sh[t+off];
    __syncthreads();
  }
  float inv = 1.0f / sh[0];
#pragma unroll
  for (int i = 0; i < 4; i++) {
    int c = t + i*256;
    probs[r * (size_t)T_ + c] = f2bf(v[i] * inv);
  }
}

// ---------------------------------------------------------------------------
// Embedding: x[bt,:] = tok_emb[idx[bt],:] + pos_emb[bt%T,:]
// ---------------------------------------------------------------------------
__global__ __launch_bounds__(256)
void embed_k(const int* __restrict__ idx, const float* __restrict__ tok,
             const float* __restrict__ pos, float* __restrict__ x)
{
  int bt = blockIdx.x;
  int tp = bt % T_;
  int id = idx[bt];
  int t = threadIdx.x;
#pragma unroll
  for (int i = 0; i < 4; i++) {
    int c = t + i*256;
    x[(size_t)bt * D_ + c] = tok[(size_t)id * D_ + c] + pos[(size_t)tp * D_ + c];
  }
}

// ---------------------------------------------------------------------------
// f32 -> bf16 conversions (weights), with row padding for head_w.
// ---------------------------------------------------------------------------
__global__ void cvt_bf16_k(const float* __restrict__ in, uint16_t* __restrict__ out,
                           size_t n)
{
  size_t stride = (size_t)gridDim.x * blockDim.x;
  for (size_t i = (size_t)blockIdx.x * blockDim.x + threadIdx.x; i < n; i += stride)
    out[i] = f2bf(in[i]);
}

__global__ void cvt_pad_k(const float* __restrict__ in, uint16_t* __restrict__ out,
                          int rows, int cols, int ldo)
{
  size_t n = (size_t)rows * cols;
  size_t stride = (size_t)gridDim.x * blockDim.x;
  for (size_t i = (size_t)blockIdx.x * blockDim.x + threadIdx.x; i < n; i += stride) {
    size_t rr = i / cols, cc = i % cols;
    out[rr * ldo + cc] = f2bf(in[i]);
  }
}

// ---------------------------------------------------------------------------
extern "C" void kernel_launch(void* const* d_in, const int* in_sizes, int n_in,
                              void* d_out, int out_size, void* d_ws, size_t ws_size,
                              hipStream_t stream)
{
  const int*   idx   = (const int*)  d_in[0];
  const float* masks = (const float*)d_in[1];
  const float* tok   = (const float*)d_in[2];
  const float* pos   = (const float*)d_in[3];
  const float* Wq = (const float*)d_in[4];   const float* bq = (const float*)d_in[5];
  const float* Wk = (const float*)d_in[6];   const float* bk = (const float*)d_in[7];
  const float* Wv = (const float*)d_in[8];   const float* bv = (const float*)d_in[9];
  const float* Wo = (const float*)d_in[10];  const float* bo = (const float*)d_in[11];
  const float* ln1w = (const float*)d_in[12]; const float* ln1b = (const float*)d_in[13];
  const float* ln2w = (const float*)d_in[14]; const float* ln2b = (const float*)d_in[15];
  const float* W1 = (const float*)d_in[16];  const float* b1 = (const float*)d_in[17];
  const float* W2 = (const float*)d_in[18];  const float* b2 = (const float*)d_in[19];
  const float* lnfw = (const float*)d_in[20]; const float* lnfb = (const float*)d_in[21];
  const float* headw = (const float*)d_in[22];
  float* out = (float*)d_out;

  // ---- workspace carve-up (256B aligned) ----
  char* ws = (char*)d_ws;
  size_t off = 0;
  auto alloc = [&](size_t bytes) -> void* {
    void* p = ws + off;
    off += (bytes + 255) & ~(size_t)255;
    return p;
  };
  uint16_t* wq_bf  = (uint16_t*)alloc((size_t)L_*D_*D_*2);
  uint16_t* wk_bf  = (uint16_t*)alloc((size_t)L_*D_*D_*2);
  uint16_t* wv_bf  = (uint16_t*)alloc((size_t)L_*D_*D_*2);
  uint16_t* wo_bf  = (uint16_t*)alloc((size_t)L_*D_*D_*2);
  uint16_t* w1_bf  = (uint16_t*)alloc((size_t)L_*D_*DFF_*2);
  uint16_t* w2_bf  = (uint16_t*)alloc((size_t)L_*DFF_*D_*2);
  uint16_t* head_bf= (uint16_t*)alloc((size_t)D_*VPAD_*2);
  float*    x      = (float*)   alloc((size_t)BT_*D_*4);
  uint16_t* h_bf   = (uint16_t*)alloc((size_t)BT_*D_*2);
  uint16_t* q_bf   = (uint16_t*)alloc((size_t)BT_*D_*2);
  uint16_t* k_bf   = (uint16_t*)alloc((size_t)BT_*D_*2);
  uint16_t* v_bf   = (uint16_t*)alloc((size_t)BT_*D_*2);
  uint16_t* ctx_bf = (uint16_t*)alloc((size_t)BT_*D_*2);
  float*    scores = (float*)   alloc((size_t)B_*H_*T_*T_*4);
  uint16_t* probs  = (uint16_t*)alloc((size_t)B_*H_*T_*T_*2);
  uint16_t* ff_bf  = (uint16_t*)alloc((size_t)BT_*DFF_*2);
  (void)ws_size; (void)in_sizes; (void)n_in; (void)out_size;

  // ---- weight conversion (every call; deterministic) ----
  cvt_bf16_k<<<2048, 256, 0, stream>>>(Wq, wq_bf, (size_t)L_*D_*D_);
  cvt_bf16_k<<<2048, 256, 0, stream>>>(Wk, wk_bf, (size_t)L_*D_*D_);
  cvt_bf16_k<<<2048, 256, 0, stream>>>(Wv, wv_bf, (size_t)L_*D_*D_);
  cvt_bf16_k<<<2048, 256, 0, stream>>>(Wo, wo_bf, (size_t)L_*D_*D_);
  cvt_bf16_k<<<4096, 256, 0, stream>>>(W1, w1_bf, (size_t)L_*D_*DFF_);
  cvt_bf16_k<<<4096, 256, 0, stream>>>(W2, w2_bf, (size_t)L_*DFF_*D_);
  cvt_pad_k <<<4096, 256, 0, stream>>>(headw, head_bf, D_, V_, VPAD_);

  embed_k<<<BT_, 256, 0, stream>>>(idx, tok, pos, x);

  auto gemmNN = [&](const uint16_t* A, const uint16_t* Bp, const float* bias,
                    const float* resid, float* Cf, uint16_t* Cb,
                    int M, int N, int K, int lda, int ldb, int ldc,
                    long long sAb, long long sAh, long long sBb, long long sBh,
                    long long sCb, long long sCh, int Hdim, int nbat, int gelu) {
    dim3 g((N + 127) / 128, M / 256, nbat);
    gemm_bf16_k<false><<<g, 256, 0, stream>>>(A, Bp, bias, resid, Cf, Cb,
        M, N, K, lda, ldb, ldc, sAb, sAh, sBb, sBh, sCb, sCh, Hdim, gelu);
  };
  auto gemmNT = [&](const uint16_t* A, const uint16_t* Bp, const float* bias,
                    const float* resid, float* Cf, uint16_t* Cb,
                    int M, int N, int K, int lda, int ldb, int ldc,
                    long long sAb, long long sAh, long long sBb, long long sBh,
                    long long sCb, long long sCh, int Hdim, int nbat, int gelu) {
    dim3 g((N + 127) / 128, M / 256, nbat);
    gemm_bf16_k<true><<<g, 256, 0, stream>>>(A, Bp, bias, resid, Cf, Cb,
        M, N, K, lda, ldb, ldc, sAb, sAh, sBb, sBh, sCb, sCh, Hdim, gelu);
  };

  const float scale = 0.125f;  // 1/sqrt(DH)

  for (int l = 0; l < L_; l++) {
    for (int m = 0; m < NM_; m++) {
      layernorm_k<<<BT_, 256, 0, stream>>>(x, ln1w + l*D_, ln1b + l*D_, h_bf);
      // Q, K, V projections
      gemmNN(h_bf, wq_bf + (size_t)l*D_*D_, bq + l*D_, nullptr, nullptr, q_bf,
             BT_, D_, D_, D_, D_, D_, 0,0,0,0,0,0, 1, 1, 0);
      gemmNN(h_bf, wk_bf + (size_t)l*D_*D_, bk + l*D_, nullptr, nullptr, k_bf,
             BT_, D_, D_, D_, D_, D_, 0,0,0,0,0,0, 1, 1, 0);
      gemmNN(h_bf, wv_bf + (size_t)l*D_*D_, bv + l*D_, nullptr, nullptr, v_bf,
             BT_, D_, D_, D_, D_, D_, 0,0,0,0,0,0, 1, 1, 0);
      // scores[b,h] = q[b,:,h*64:] @ k[b,:,h*64:]^T   (batched over b*h)
      gemmNT(q_bf, k_bf, nullptr, nullptr, scores, nullptr,
             T_, T_, DH_, D_, D_, T_,
             (long long)T_*D_, DH_, (long long)T_*D_, DH_,
             (long long)H_*T_*T_, (long long)T_*T_, H_, B_*H_, 0);
      softmax_mask_k<<<B_*H_*T_, 256, 0, stream>>>(
          scores, masks + (size_t)m*T_*T_, probs, scale);
      // ctx[b,:,h*64:] = probs[b,h] @ v[b,:,h*64:]
      gemmNN(probs, v_bf, nullptr, nullptr, nullptr, ctx_bf,
             T_, DH_, T_, T_, D_, D_,
             (long long)H_*T_*T_, (long long)T_*T_, (long long)T_*D_, DH_,
             (long long)T_*D_, DH_, H_, B_*H_, 0);
      // x += ctx @ Wo + bo
      gemmNN(ctx_bf, wo_bf + (size_t)l*D_*D_, bo + l*D_, x, x, nullptr,
             BT_, D_, D_, D_, D_, D_, 0,0,0,0,0,0, 1, 1, 0);
    }
    // MLP
    layernorm_k<<<BT_, 256, 0, stream>>>(x, ln2w + l*D_, ln2b + l*D_, h_bf);
    gemmNN(h_bf, w1_bf + (size_t)l*D_*DFF_, b1 + l*DFF_, nullptr, nullptr, ff_bf,
           BT_, DFF_, D_, D_, DFF_, DFF_, 0,0,0,0,0,0, 1, 1, 1);
    gemmNN(ff_bf, w2_bf + (size_t)l*DFF_*D_, b2 + l*D_, x, x, nullptr,
           BT_, D_, DFF_, DFF_, D_, D_, 0,0,0,0,0,0, 1, 1, 0);
  }

  // final LN + vocab head
  layernorm_k<<<BT_, 256, 0, stream>>>(x, lnfw, lnfb, h_bf);
  gemmNN(h_bf, head_bf, nullptr, nullptr, out, nullptr,
         BT_, V_, D_, D_, VPAD_, V_, 0,0,0,0,0,0, 1, 1, 0);
}
